// DConv_89386859364430
// MI455X (gfx1250) — matrix operand order repaired and checked by reading, not compiled, not run on hardware
//
#include <hip/hip_runtime.h>

typedef __attribute__((ext_vector_type(16))) __bf16 v16bf;
typedef __attribute__((ext_vector_type(8)))  __bf16 v8bf;
typedef __attribute__((ext_vector_type(8)))  float  v8f;
typedef __attribute__((ext_vector_type(4)))  unsigned int v4u;
typedef __attribute__((ext_vector_type(8)))  int  v8i;
typedef __attribute__((ext_vector_type(4)))  int  v4i;

#define N_     8
#define C_     256
#define H_     64
#define W_     64
#define KK_    9
#define COUT_  256
#define OFFCH  18
#define HW_    (H_ * W_)          // 4096
#define CHW_   (C_ * HW_)         // 1048576
#define M_TOTAL (N_ * HW_)        // 32768
#define KDIM   (C_ * KK_)         // 2304  (tap-major: k' = kk*256 + c)

#define M_TILE 128
#define KB     32
#define NSLAB  (KDIM / KB)        // 72 ; 8 slabs per tap
#define LDA    40                 // bf16 elems per A row (80 B, 16B-aligned)
#define LDB    40                 // bf16 elems per B column (K-major, 64B+16B pad)

#define USE_TDM_B   1             // TENSOR_LOAD_TO_LDS for B staging (1 DMA/slab)
#define USE_ASYNC_B 1             // fallback: GLOBAL_LOAD_ASYNC_TO_LDS_B128

// ---------------------------------------------------------------------------
// Kernel 1: offset-predicting conv  (x * w1 -> off[8,18,64,64])
// ---------------------------------------------------------------------------
__global__ __launch_bounds__(256) void offset_conv_kernel(
    const float* __restrict__ x, const float* __restrict__ w1,
    float* __restrict__ off) {
  int m = blockIdx.x * blockDim.x + threadIdx.x;   // 0..32767
  int w = m & (W_ - 1);
  int h = (m >> 6) & (H_ - 1);
  int n = m >> 12;
  float acc[OFFCH];
#pragma unroll
  for (int i = 0; i < OFFCH; ++i) acc[i] = 0.f;
  const float* xn = x + (size_t)n * CHW_;
  for (int c = 0; c < C_; ++c) {
    float xv[9];
#pragma unroll
    for (int t = 0; t < 9; ++t) {
      int hy = h - 1 + t / 3;
      int wx = w - 1 + t % 3;
      bool ok = (hy >= 0) & (hy < H_) & (wx >= 0) & (wx < W_);
      xv[t] = ok ? xn[c * HW_ + hy * W_ + wx] : 0.f;
    }
#pragma unroll
    for (int co = 0; co < OFFCH; ++co) {
      const float* wk = w1 + ((size_t)co * C_ + c) * 9;   // uniform -> s_load
      float s = 0.f;
#pragma unroll
      for (int t = 0; t < 9; ++t) s += wk[t] * xv[t];
      acc[co] += s;
    }
  }
#pragma unroll
  for (int co = 0; co < OFFCH; ++co)
    off[(((size_t)n * OFFCH + co) * H_ + h) * W_ + w] = acc[co];
}

// ---------------------------------------------------------------------------
// Kernel 2: w2 [Cout][c*9+kk] f32 -> Bg [Cout][k'=kk*256+c] bf16 (co-major)
// ---------------------------------------------------------------------------
__global__ __launch_bounds__(256) void w2_to_bf16_kernel(
    const float* __restrict__ w2, __bf16* __restrict__ Bg) {
  int idx = blockIdx.x * blockDim.x + threadIdx.x;
  if (idx >= COUT_ * KDIM) return;
  int co = idx / KDIM;
  int k  = idx - co * KDIM;      // c*9 + kk
  int c  = k / 9;
  int kk = k - c * 9;
  int kp = kk * C_ + c;          // tap-major K index
  Bg[(size_t)co * KDIM + kp] = (__bf16)w2[idx];
}

// ---------------------------------------------------------------------------
// Bilinear sampling params for (row, tap): clamped corner indices + masked wgts
// ---------------------------------------------------------------------------
__device__ __forceinline__ void samp_params(const float* __restrict__ off,
                                            int n, int h, int w, int kk,
                                            int* idx4, float* wgt4) {
  float dy = off[(((size_t)n * OFFCH + 2 * kk) * H_ + h) * W_ + w];
  float dx = off[(((size_t)n * OFFCH + 2 * kk + 1) * H_ + h) * W_ + w];
  float py = (float)(h - 1 + kk / 3) + dy;
  float px = (float)(w - 1 + kk % 3) + dx;
  float fy = floorf(py), fx = floorf(px);
  float wy = py - fy, wx = px - fx;
  int y0 = (int)fy, x0 = (int)fx;
  float ww[4] = {(1.f - wy) * (1.f - wx), (1.f - wy) * wx,
                 wy * (1.f - wx),         wy * wx};
  int yy[4] = {y0, y0, y0 + 1, y0 + 1};
  int xx[4] = {x0, x0 + 1, x0, x0 + 1};
#pragma unroll
  for (int i = 0; i < 4; ++i) {
    bool ok = (yy[i] >= 0) & (yy[i] < H_) & (xx[i] >= 0) & (xx[i] < W_);
    int yc = min(max(yy[i], 0), H_ - 1);
    int xc = min(max(xx[i], 0), W_ - 1);
    idx4[i] = yc * W_ + xc;            // always valid -> unconditional load
    wgt4[i] = ok ? ww[i] : 0.f;        // zero weight handles out-of-bounds
  }
}

// ---------------------------------------------------------------------------
// Kernel 3: fused bilinear sampling + implicit GEMM (double-buffered LDS)
// 256 blocks (M tiles of 128), 256 threads = 8 waves.
// Wave wv: 8 M-frags x 2 N-frags (cols [wv*32, wv*32+32)).
// ---------------------------------------------------------------------------
__global__ __launch_bounds__(256) void deform_gemm_kernel(
    const float* __restrict__ x, const float* __restrict__ off,
    const __bf16* __restrict__ Bg, const float* __restrict__ b2,
    float* __restrict__ out) {
  __shared__ __align__(16) __bf16 sA[2][M_TILE * LDA];   // 2 x 10240 B
  __shared__ __align__(16) __bf16 sB[2][COUT_ * LDB];    // 2 x 20480 B

  const int tid   = threadIdx.x;
  const int mBase = blockIdx.x * M_TILE;
  const int wv    = tid >> 5;
  const int lane  = tid & 31;
  const int half  = lane >> 4;
  const int l16   = lane & 15;

  // staging identity: this thread stages row r, contiguous K-phase kh
  const int r  = tid & 127;
  const int kh = tid >> 7;              // kq = kh*16 + j, j = 0..15
  const int rm = mBase + r;
  const int rw = rm & (W_ - 1);
  const int rh = (rm >> 6) & (H_ - 1);
  const int rn = rm >> 12;
  const float* xb = x + (size_t)rn * CHW_ + (size_t)kh * 16 * HW_;

  int   idx4[4];
  float wgt4[4];

  // stage slab k0 (tap params must already match k0>>8) into buffer sel
  auto stage = [&](int k0, int sel) {
    // ---- A: 32 gathered values per row; batched loads, 2x b128 LDS store ----
    const float* xc0 = xb + (k0 & 255) * HW_;
    float fv[16];
#pragma unroll
    for (int jb = 0; jb < 16; jb += 4) {
      float t[16];                       // 16 independent loads in flight
#pragma unroll
      for (int j2 = 0; j2 < 4; ++j2) {
        const float* xc = xc0 + (jb + j2) * HW_;
#pragma unroll
        for (int i = 0; i < 4; ++i) t[j2 * 4 + i] = xc[idx4[i]];
      }
#pragma unroll
      for (int j2 = 0; j2 < 4; ++j2)
        fv[jb + j2] = wgt4[0] * t[j2 * 4 + 0] + wgt4[1] * t[j2 * 4 + 1] +
                      wgt4[2] * t[j2 * 4 + 2] + wgt4[3] * t[j2 * 4 + 3];
    }
    v8bf p0, p1;
#pragma unroll
    for (int e = 0; e < 8; ++e) {
      p0[e] = (__bf16)fv[e];
      p1[e] = (__bf16)fv[8 + e];
    }
    *(v8bf*)&sA[sel][r * LDA + kh * 16]     = p0;
    *(v8bf*)&sA[sel][r * LDA + kh * 16 + 8] = p1;

    // ---- B: one TDM descriptor moves the whole 256x32 bf16 tile, with the
    //      16B-per-64B LDS padding producing the LDB=40 column layout. ----
#if USE_TDM_B
    if (tid < 32) {                      // wave 0 only: TDM ignores EXEC
      unsigned lds = (unsigned)(uintptr_t)&sB[sel][0];
      unsigned long long ga = (unsigned long long)(uintptr_t)(Bg + k0);
      v4u g0;
      g0[0] = 1u;                                          // count=1, user D#
      g0[1] = lds;                                         // lds_addr
      g0[2] = (unsigned)(ga & 0xffffffffu);                // global_addr[31:0]
      g0[3] = (unsigned)((ga >> 32) & 0x1ffffffu) | (2u << 30); // [56:32]|type=2
      v8i g1;
      g1[0] = (1 << 16)            // data_size = 2 bytes
            | (1 << 20)            // pad_enable
            | (3 << 22)            // pad_interval: every 16 dwords (64 B)
            | (3 << 25);           // pad_amount: 4 dwords (16 B)
      g1[1] = (int)((KDIM & 0xffff) << 16);                       // dim0 lo
      g1[2] = (int)(((KDIM >> 16) & 0xffff) | ((COUT_ & 0xffff) << 16)); // dim0 hi | dim1 lo
      g1[3] = (int)(((COUT_ >> 16) & 0xffff) | (KB << 16));       // dim1 hi | tile_dim0=32
      g1[4] = COUT_;                                              // tile_dim1=256, tile_dim2=0
      g1[5] = KDIM;                                               // dim0_stride lo
      g1[6] = 0;                                                  // dim0_stride hi | dim1_stride lo
      g1[7] = 0;
      v4i gz4 = (v4i)0;
      v8i gz8 = (v8i)0;
      __builtin_amdgcn_tensor_load_to_lds(g0, g1, gz4, gz4, gz8, 0);
    }
#elif USE_ASYNC_B
#pragma unroll
    for (int q = 0; q < 4; ++q) {
      unsigned la = (unsigned)(uintptr_t)&sB[sel][tid * LDB + q * 8];
      unsigned long long ga =
          (unsigned long long)(uintptr_t)(Bg + (size_t)tid * KDIM + k0 + q * 8);
      asm volatile("global_load_async_to_lds_b128 %0, %1, off"
                   :: "v"(la), "v"(ga) : "memory");
    }
#else
    v8bf bt[4];
#pragma unroll
    for (int q = 0; q < 4; ++q)
      bt[q] = *(const v8bf*)(Bg + (size_t)tid * KDIM + k0 + q * 8);
#pragma unroll
    for (int q = 0; q < 4; ++q)
      *(v8bf*)&sB[sel][tid * LDB + q * 8] = bt[q];
#endif
  };

  // drain outstanding B staging before the workgroup barrier
  auto stage_fence = [&]() {
#if USE_TDM_B
    if (tid < 32) __builtin_amdgcn_s_wait_tensorcnt(0);
#elif USE_ASYNC_B
    asm volatile("s_wait_asynccnt 0x0" ::: "memory");
#endif
  };

  v8f acc[8][2];
#pragma unroll
  for (int mf = 0; mf < 8; ++mf)
#pragma unroll
    for (int nf = 0; nf < 2; ++nf) acc[mf][nf] = (v8f){};

  samp_params(off, rn, rh, rw, 0, idx4, wgt4);
  stage(0, 0);
  stage_fence();
  __syncthreads();

#pragma unroll 2
  for (int s = 0; s < NSLAB; ++s) {
    const int sel = s & 1;
    // prefetch next slab into the other buffer (overlaps WMMA below)
    if (s + 1 < NSLAB) {
      if (((s + 1) & 7) == 0)
        samp_params(off, rn, rh, rw, (s + 1) >> 3, idx4, wgt4);
      stage((s + 1) * KB, sel ^ 1);
    }

    const __bf16* aBuf = sA[sel];
    const __bf16* bBuf = sB[sel];

    // B fragments (32x16): lanes 0-15 K=0..15, lanes 16-31 K=16..31
    v16bf bfrag[2];
#pragma unroll
    for (int nf = 0; nf < 2; ++nf) {
      int ncol = wv * 32 + nf * 16 + l16;
      v8bf lo = *(const v8bf*)&bBuf[ncol * LDB + half * 16];
      v8bf hi = *(const v8bf*)&bBuf[ncol * LDB + half * 16 + 8];
      bfrag[nf] = __builtin_shufflevector(lo, hi, 0, 1, 2, 3, 4, 5, 6, 7,
                                          8, 9, 10, 11, 12, 13, 14, 15);
    }
    // A fragments (16x32 contiguous-half layout: lanes 0-15 K=0..15, 16-31 K=16..31) + WMMA
#pragma unroll
    for (int mf = 0; mf < 8; ++mf) {
      int mrow = mf * 16 + l16;
      v8bf lo = *(const v8bf*)&aBuf[mrow * LDA + half * 16];
      v8bf hi = *(const v8bf*)&aBuf[mrow * LDA + half * 16 + 8];
      v16bf afrag = __builtin_shufflevector(lo, hi, 0, 1, 2, 3, 4, 5, 6, 7,
                                            8, 9, 10, 11, 12, 13, 14, 15);
      acc[mf][0] = __builtin_amdgcn_wmma_f32_16x16x32_bf16(
          false, afrag, false, bfrag[0], (short)0, acc[mf][0], false, false);
      acc[mf][1] = __builtin_amdgcn_wmma_f32_16x16x32_bf16(
          false, afrag, false, bfrag[1], (short)0, acc[mf][1], false, false);
    }
    stage_fence();
    __syncthreads();
  }

  // Epilogue: D layout: VGPR e, lanes0-15 -> M=e, lanes16-31 -> M=8+e
#pragma unroll
  for (int mf = 0; mf < 8; ++mf) {
    int mm0 = mBase + mf * 16 + half * 8;  // multiple of 8 -> row-contiguous
    int w0 = mm0 & (W_ - 1);
    int h  = (mm0 >> 6) & (H_ - 1);
    int n  = mm0 >> 12;
#pragma unroll
    for (int nf = 0; nf < 2; ++nf) {
      int co = wv * 32 + nf * 16 + l16;
      float bias = b2[co];
      v8f o = acc[mf][nf];
#pragma unroll
      for (int e = 0; e < 8; ++e) o[e] += bias;
      *(v8f*)&out[(((size_t)n * COUT_ + co) * H_ + h) * W_ + w0] = o;
    }
  }
}

// ---------------------------------------------------------------------------
extern "C" void kernel_launch(void* const* d_in, const int* in_sizes, int n_in,
                              void* d_out, int out_size, void* d_ws, size_t ws_size,
                              hipStream_t stream) {
  const float* x  = (const float*)d_in[0];
  const float* w1 = (const float*)d_in[1];
  const float* w2 = (const float*)d_in[2];
  const float* b2 = (const float*)d_in[3];
  float* out = (float*)d_out;

  float*  off = (float*)d_ws;                                        // 2.36 MB
  __bf16* Bg  = (__bf16*)((char*)d_ws +
                          (size_t)N_ * OFFCH * HW_ * sizeof(float)); // 1.18 MB

  offset_conv_kernel<<<M_TOTAL / 256, 256, 0, stream>>>(x, w1, off);
  w2_to_bf16_kernel<<<(COUT_ * KDIM + 255) / 256, 256, 0, stream>>>(w2, Bg);
  deform_gemm_kernel<<<M_TOTAL / M_TILE, 256, 0, stream>>>(x, off, Bg, b2, out);
}
